// GCNII_35321811042823
// MI455X (gfx1250) — matrix operand order repaired and checked by reading, not compiled, not run on hardware
//
#include <hip/hip_runtime.h>
#include <hip/hip_bf16.h>
#include <math.h>

// GCNII on MI455X (gfx1250, wave32).
//  - SpMM: wave-per-edge, lane = feature (F=32 == wave32). Edge arrays (192MB
//    per layer) streamed with non-temporal loads so the 64MB h and 64MB agg
//    matrices stay resident in the 192MB L2; gathers/atomic scatters are
//    coalesced 128B transactions served from L2.
//  - Dense transform: fused blend (0.9*agg + 0.1*x0@w_init) -> f16 LDS staging
//    -> v_wmma_f32_16x16x32_f16 (K=32 == hidden dim) -> ReLU -> store.
//    Wave index scalarized via readfirstlane so tile guards compile to SALU
//    branches and the full-tile store path is branch-free.
//  - Layer 0 (F0=3) and last layer ([32,1]+sigmoid) are skinny: VALU kernels,
//    last layer uses a wave32 shuffle reduction.

typedef __attribute__((ext_vector_type(16))) _Float16 v16h;
typedef __attribute__((ext_vector_type(8)))  float    v8f;

#define GCN_ALPHA 0.9f
#define GCN_BETA  0.1f

// ---------------- SpMM: F=3 (first layer aggregation) ----------------
__global__ void spmm_f3_kernel(const int* __restrict__ rows,
                               const int* __restrict__ cols,
                               const float* __restrict__ vals,
                               const float* __restrict__ x,
                               float* __restrict__ agg, int E) {
  int stride = gridDim.x * blockDim.x;
  for (int e = blockIdx.x * blockDim.x + threadIdx.x; e < E; e += stride) {
    int   r = __builtin_nontemporal_load(&rows[e]);
    int   c = __builtin_nontemporal_load(&cols[e]);
    float v = __builtin_nontemporal_load(&vals[e]);
    const float* xs = x + (size_t)c * 3;
    float*       as = agg + (size_t)r * 3;
    atomicAdd(&as[0], v * xs[0]);
    atomicAdd(&as[1], v * xs[1]);
    atomicAdd(&as[2], v * xs[2]);
  }
}

// ---------------- SpMM: F=32, wave-per-edge, lane = feature ----------------
__global__ void spmm_f32_kernel(const int* __restrict__ rows,
                                const int* __restrict__ cols,
                                const float* __restrict__ vals,
                                const float* __restrict__ h,
                                float* __restrict__ agg, int E) {
  int lane = threadIdx.x & 31;
  int wid  = blockIdx.x * (blockDim.x >> 5) + (threadIdx.x >> 5);
  int nw   = gridDim.x * (blockDim.x >> 5);
  for (int e = wid; e < E; e += nw) {
    int   r = __builtin_nontemporal_load(&rows[e]);   // NT: keep L2 for h/agg
    int   c = __builtin_nontemporal_load(&cols[e]);
    float v = __builtin_nontemporal_load(&vals[e]);
    float g = h[(size_t)c * 32 + lane];               // coalesced, L2-resident
    atomicAdd(&agg[(size_t)r * 32 + lane], v * g);    // coalesced L2 atomic
  }
}

// ---------------- Layer 0: blend(F=3) @ w_x0[3,32] + ReLU ----------------
__global__ void layer0_kernel(const float* __restrict__ agg0,
                              const float* __restrict__ x,
                              const float* __restrict__ w_init0, // [3][3]
                              const float* __restrict__ w_x0,    // [3][32]
                              float* __restrict__ hout, int n) {
  size_t t = (size_t)blockIdx.x * blockDim.x + threadIdx.x;
  if (t >= (size_t)n * 32) return;
  int node = (int)(t >> 5);
  int j    = (int)(t & 31);
  const float* xp = x    + (size_t)node * 3;
  const float* ap = agg0 + (size_t)node * 3;
  float bl[3];
#pragma unroll
  for (int f = 0; f < 3; ++f) {
    float xw = xp[0] * w_init0[0 * 3 + f] + xp[1] * w_init0[1 * 3 + f] +
               xp[2] * w_init0[2 * 3 + f];
    bl[f] = GCN_ALPHA * ap[f] + GCN_BETA * xw;
  }
  float acc = bl[0] * w_x0[0 * 32 + j] + bl[1] * w_x0[1 * 32 + j] +
              bl[2] * w_x0[2 * 32 + j];
  hout[t] = fmaxf(acc, 0.f);
}

// ------- Mid layers: blend + [N,32]@[32,32] via v_wmma_f32_16x16x32_f16 ------
__global__ __launch_bounds__(256)
void blend_gemm_relu_kernel(const float* __restrict__ agg,
                            const float* __restrict__ x,
                            const float* __restrict__ w_init, // [3][32]
                            const float* __restrict__ w_x,    // [32][32]
                            float* __restrict__ hout, int n) {
  // Per-wave A tile staging + transposed f16 B tile. Rows padded to 40 halves
  // (80B) so 16 tile rows start on distinct LDS bank groups.
  __shared__ _Float16 sA[8][16][40];
  __shared__ _Float16 sBt[32][40];   // sBt[ncol][k] = (f16) w_x[k][ncol]

  int tid  = threadIdx.x;
  int lane = tid & 31;
  // Wave index is uniform across the wave: scalarize it so all tile guards
  // below compile to SALU compares / s_cbranch instead of exec masking.
  int wv   = __builtin_amdgcn_readfirstlane(tid >> 5);

  // Cooperative transpose+convert of w_x (1024 elements, 256 threads).
  for (int i = tid; i < 1024; i += 256) {
    int k = i & 31, ncol = i >> 5;
    sBt[ncol][k] = (_Float16)w_x[k * 32 + ncol];
  }

  int  tile   = blockIdx.x * 8 + wv;
  int  base   = tile * 16;
  bool active = base < n;            // scalar

  if (active) {
    // lane = output feature column j; build blended f16 A tile rows 0..15
    int j = lane;
    float wi0 = w_init[j], wi1 = w_init[32 + j], wi2 = w_init[64 + j];
#pragma unroll
    for (int r = 0; r < 16; ++r) {
      int node = base + r;           // scalar predicate
      float b = 0.f;
      if (node < n) {
        float a = agg[(size_t)node * 32 + j];
        const float* xp = x + (size_t)node * 3;
        b = GCN_ALPHA * a + GCN_BETA * (xp[0] * wi0 + xp[1] * wi1 + xp[2] * wi2);
      }
      sA[wv][r][j] = (_Float16)b;
    }
  }
  __syncthreads();
  if (!active) return;               // scalar exit; EXEC stays all-1s

  // A fragment (this code's contiguous k order, same as B): lanes 0-15 hold K=0..15 of row
  // M=lane; lanes 16-31 hold K=16..31 of row M=lane-16.
  int m  = lane & 15;
  int k0 = (lane >> 4) * 16;
  v16h a;
#pragma unroll
  for (int e = 0; e < 8; ++e) a[e] = sA[wv][m][k0 + e];
#pragma unroll
  for (int e = 0; e < 8; ++e) a[8 + e] = sA[wv][m][k0 + 8 + e];

  // B fragment (same contiguous k order as A): lanes 0-15 hold K=0..15 of column N=lane,
  // lanes 16-31 hold K=16..31 of column N=lane-16.
  int kb0 = (lane >> 4) * 16;
  v8f cacc[2];
#pragma unroll
  for (int jt = 0; jt < 2; ++jt) {
    v16h b;
#pragma unroll
    for (int e = 0; e < 16; ++e) b[e] = sBt[jt * 16 + m][kb0 + e];
    v8f c = {};
    cacc[jt] = __builtin_amdgcn_wmma_f32_16x16x32_f16(
        /*neg_a=*/false, a, /*neg_b=*/false, b,
        /*c_mod=*/(short)0, c, /*reuse_a=*/false, /*reuse_b=*/false);
  }

  // C/D layout: VGPR r -> row M=r (lanes 0-15) / M=r+8 (lanes 16-31), col=lane&15.
  int rowOff = (lane >> 4) * 8;
  if (base + 16 <= n) {
    // Full tile: branch-free coalesced stores.
#pragma unroll
    for (int jt = 0; jt < 2; ++jt)
#pragma unroll
      for (int r = 0; r < 8; ++r)
        hout[(size_t)(base + rowOff + r) * 32 + jt * 16 + m] =
            fmaxf(cacc[jt][r], 0.f);
  } else {
#pragma unroll
    for (int jt = 0; jt < 2; ++jt)
#pragma unroll
      for (int r = 0; r < 8; ++r) {
        int node = base + rowOff + r;
        if (node < n)
          hout[(size_t)node * 32 + jt * 16 + m] = fmaxf(cacc[jt][r], 0.f);
      }
  }
}

// ------- Last layer: blend -> dot w_x[32,1] -> sigmoid (wave per node) -------
__global__ void last_layer_kernel(const float* __restrict__ agg,
                                  const float* __restrict__ x,
                                  const float* __restrict__ w_init, // [3][32]
                                  const float* __restrict__ w_x,    // [32][1]
                                  float* __restrict__ out, int n) {
  int lane = threadIdx.x & 31;
  int node = blockIdx.x * (blockDim.x >> 5) + (threadIdx.x >> 5);
  if (node >= n) return;
  const float* xp = x + (size_t)node * 3;
  float b = GCN_ALPHA * agg[(size_t)node * 32 + lane] +
            GCN_BETA * (xp[0] * w_init[lane] + xp[1] * w_init[32 + lane] +
                        xp[2] * w_init[64 + lane]);
  float p = b * w_x[lane];
#pragma unroll
  for (int off = 16; off; off >>= 1) p += __shfl_xor(p, off, 32);
  if (lane == 0) out[node] = 1.f / (1.f + __expf(-p));
}

extern "C" void kernel_launch(void* const* d_in, const int* in_sizes, int n_in,
                              void* d_out, int out_size, void* d_ws, size_t ws_size,
                              hipStream_t stream) {
  const float* x           = (const float*)d_in[0];
  const int*   rows        = (const int*)d_in[1];
  const int*   cols        = (const int*)d_in[2];
  const float* vals        = (const float*)d_in[3];
  const float* w_init0     = (const float*)d_in[4];   // [3,3]
  const float* w_x0        = (const float*)d_in[5];   // [3,32]
  const float* w_init_mid  = (const float*)d_in[6];   // [6,3,32]
  const float* w_x_mid     = (const float*)d_in[7];   // [6,32,32]
  const float* w_init_last = (const float*)d_in[8];   // [3,32]
  const float* w_x_last    = (const float*)d_in[9];   // [32,1]

  int n = in_sizes[0] / 3;
  int E = in_sizes[1];

  // Workspace: h[N,32] | agg[N,32] | agg0[N,3]  (~134 MB for N=500k)
  float* h    = (float*)d_ws;
  float* agg  = h + (size_t)n * 32;
  float* agg0 = agg + (size_t)n * 32;

  const dim3 blk(256);
  const int spmmBlocks = 3072;                       // 24576 waves over 16M edges
  const int tiles      = (n + 15) / 16;
  const int gemmBlocks = (tiles + 7) / 8;            // 8 waves (tiles) per block
  const int waveBlocks = (n + 7) / 8;                // 8 waves per 256-thread block

  // ---- Layer 0 ----
  hipMemsetAsync(agg0, 0, (size_t)n * 3 * sizeof(float), stream);
  spmm_f3_kernel<<<spmmBlocks, blk, 0, stream>>>(rows, cols, vals, x, agg0, E);
  {
    size_t total = (size_t)n * 32;
    int g = (int)((total + 255) / 256);
    layer0_kernel<<<g, blk, 0, stream>>>(agg0, x, w_init0, w_x0, h, n);
  }

  // ---- 6 mid layers ----
  for (int i = 0; i < 6; ++i) {
    hipMemsetAsync(agg, 0, (size_t)n * 32 * sizeof(float), stream);
    spmm_f32_kernel<<<spmmBlocks, blk, 0, stream>>>(rows, cols, vals, h, agg, E);
    blend_gemm_relu_kernel<<<gemmBlocks, blk, 0, stream>>>(
        agg, x, w_init_mid + (size_t)i * 96, w_x_mid + (size_t)i * 1024, h, n);
  }

  // ---- Last layer ----
  hipMemsetAsync(agg, 0, (size_t)n * 32 * sizeof(float), stream);
  spmm_f32_kernel<<<spmmBlocks, blk, 0, stream>>>(rows, cols, vals, h, agg, E);
  last_layer_kernel<<<waveBlocks, blk, 0, stream>>>(agg, x, w_init_last,
                                                    w_x_last, (float*)d_out, n);
}
